// ComplexMLPS_72799695667300
// MI455X (gfx1250) — matrix operand order repaired and checked by reading, not compiled, not run on hardware
//
#include <hip/hip_runtime.h>
#include <hip/hip_bf16.h>

typedef __attribute__((ext_vector_type(16))) _Float16 v16h;
typedef __attribute__((ext_vector_type(8)))  _Float16 v8h;
typedef __attribute__((ext_vector_type(8)))  float    v8f;

#define HG_L    8
#define HG_T    (1u << 19)
#define NEUR    128
#define ENC_DIM 32
#define PTS     128    // points per block (8 waves x 16 rows)
#define THREADS 256
#define ENCS    40     // enc / W1 stride in halves (80 B, 16B-aligned, bank-skewed)
#define ACTS    136    // activation / W2 / W3 stride in halves (272 B)

// ---- WMMA fragment helpers (layouts per CDNA5 ISA 7.12.2) ----

// A: 16x32 f16 tile, rows m0..m0+15, base points at (row0, k0) of a row-major
// LDS matrix with stride_h halves. Lane l<16: K chunks {0..7}+{16..23}; lane
// l>=16: +8 on each chunk.
__device__ __forceinline__ v16h load_frag_a(const _Float16* base, int stride_h, int lane) {
    const int m  = lane & 15;
    const int hi = lane >> 4;
    const _Float16* p = base + m * stride_h + (hi << 3);
    union { v16h v; v8h h[2]; } u;
    u.h[0] = *(const v8h*)(p);        // K = 8*hi .. 8*hi+7
    u.h[1] = *(const v8h*)(p + 16);   // K = 16+8*hi .. 23+8*hi
    return u.v;
}

// B: 32x16 f16 tile with B[k][n] = W[n0+n][k0+k]; W row-major [out][in],
// stride_h halves. Lane holds column n = lane&15; K chunks {0..7}+{16..23}, +8 for hi.
__device__ __forceinline__ v16h load_frag_b(const _Float16* wbase, int stride_h, int k0, int lane) {
    const int n  = lane & 15;
    const int hi = lane >> 4;
    const _Float16* p = wbase + n * stride_h + k0 + (hi << 3);
    union { v16h v; v8h h[2]; } u;
    u.h[0] = *(const v8h*)(p);
    u.h[1] = *(const v8h*)(p + 16);
    return u.v;
}

// D/C f32 16x16: lane n = lane&15, VGPR r -> row r + 8*(lane>>4).
__device__ __forceinline__ void store_act_tile(_Float16* act, int m0, int n0,
                                               v8f c, const float* bias,
                                               bool do_relu, int lane) {
    const int n  = lane & 15;
    const int hi = lane >> 4;
    const float b = bias[n0 + n];
#pragma unroll
    for (int r = 0; r < 8; ++r) {
        float v = c[r] + b;
        if (do_relu) v = fmaxf(v, 0.0f);
        act[(m0 + r + (hi << 3)) * ACTS + n0 + n] = (_Float16)v;
    }
}

__global__ __launch_bounds__(THREADS)
void ngp_fused_mlp_kernel(const float* __restrict__ x,
                          const float* __restrict__ table,
                          const float* __restrict__ W1, const float* __restrict__ b1,
                          const float* __restrict__ W2, const float* __restrict__ b2,
                          const float* __restrict__ W3, const float* __restrict__ b3,
                          const float* __restrict__ W4, const float* __restrict__ b4,
                          float* __restrict__ out, int N) {
    __shared__ __align__(16) _Float16 s_enc [PTS * ENCS];
    __shared__ __align__(16) _Float16 s_actA[PTS * ACTS];
    __shared__ __align__(16) _Float16 s_actB[PTS * ACTS];
    __shared__ __align__(16) _Float16 s_w1  [NEUR * ENCS];
    __shared__ __align__(16) _Float16 s_w2  [NEUR * ACTS];
    __shared__ __align__(16) _Float16 s_w3  [NEUR * ACTS];
    __shared__ __align__(16) float    s_w4  [2 * NEUR];
    __shared__ float s_b1[NEUR], s_b2[NEUR], s_b3[NEUR], s_b4[2];

    const int t = threadIdx.x;

    // ---- stage 0: preload weights to LDS (f32 -> f16, padded strides) ----
    for (int i = t; i < NEUR * ENC_DIM; i += THREADS)
        s_w1[(i >> 5) * ENCS + (i & 31)] = (_Float16)W1[i];
    for (int i = t; i < NEUR * NEUR; i += THREADS) {
        const int r = i >> 7, c = i & 127;
        s_w2[r * ACTS + c] = (_Float16)W2[i];
        s_w3[r * ACTS + c] = (_Float16)W3[i];
    }
    for (int i = t; i < 2 * NEUR; i += THREADS) s_w4[i] = W4[i];
    if (t < NEUR) { s_b1[t] = b1[t]; s_b2[t] = b2[t]; s_b3[t] = b3[t]; }
    if (t < 2)    s_b4[t] = b4[t];

    // ---- stage 1: encoding into s_enc (f16, 32 features) ----
    const int pl = t & (PTS - 1);
    long pg = (long)blockIdx.x * PTS + pl;
    if (pg >= N) pg = (long)N - 1;   // clamp; keeps EXEC full for WMMA later
    _Float16* er = s_enc + pl * ENCS;

    if (t < PTS) {
        // hash-grid encode (features 0..15)
        const float* xp = x + pg * 6;
        const float px0 = xp[0], px1 = xp[1], px2 = xp[2];
#pragma unroll
        for (int l = 0; l < HG_L; ++l) {
            const float res = (float)(16 << l);
            const float fx = px0 * res, fy = px1 * res, fz = px2 * res;
            const float flx = floorf(fx), fly = floorf(fy), flz = floorf(fz);
            const unsigned ix = (unsigned)flx, iy = (unsigned)fly, iz = (unsigned)flz;
            const float wx = fx - flx, wy = fy - fly, wz = fz - flz;
            const float* tl = table + (size_t)l * (size_t)HG_T * 2u;
            float f0 = 0.0f, f1 = 0.0f;
#pragma unroll
            for (int c = 0; c < 8; ++c) {
                const unsigned cx = ix + (c & 1);
                const unsigned cy = iy + ((c >> 1) & 1);
                const unsigned cz = iz + ((c >> 2) & 1);
                const unsigned h  = cx ^ (cy * 2654435761u) ^ (cz * 805459861u);
                const unsigned idx = h & (HG_T - 1u);
                const float w = ((c & 1)        ? wx : 1.0f - wx)
                              * (((c >> 1) & 1) ? wy : 1.0f - wy)
                              * (((c >> 2) & 1) ? wz : 1.0f - wz);
                const float2 f = *(const float2*)(tl + (size_t)idx * 2u);
                f0 = fmaf(w, f.x, f0);
                f1 = fmaf(w, f.y, f1);
            }
            er[2 * l]     = (_Float16)f0;
            er[2 * l + 1] = (_Float16)f1;
        }
    } else {
        // SH degree-4 encode (features 16..31)
        const float* xp = x + pg * 6 + 3;
        const float dx = 2.0f * xp[0] - 1.0f;
        const float dy = 2.0f * xp[1] - 1.0f;
        const float dz = 2.0f * xp[2] - 1.0f;
        const float x2 = dx * dx, y2 = dy * dy, z2 = dz * dz;
        const float xy = dx * dy, yz = dy * dz, xz = dx * dz;
        float sh[16];
        sh[0]  = 0.28209479177387814f;
        sh[1]  = -0.48860251190291987f * dy;
        sh[2]  =  0.48860251190291987f * dz;
        sh[3]  = -0.48860251190291987f * dx;
        sh[4]  =  1.0925484305920792f * xy;
        sh[5]  = -1.0925484305920792f * yz;
        sh[6]  =  0.94617469575756f * z2 - 0.31539156525252f;
        sh[7]  = -1.0925484305920792f * xz;
        sh[8]  =  0.5462742152960396f * (x2 - y2);
        sh[9]  =  0.5900435899266435f * dy * (-3.0f * x2 + y2);
        sh[10] =  2.890611442640554f * xy * dz;
        sh[11] =  0.4570457994644657f * dy * (1.0f - 5.0f * z2);
        sh[12] =  0.3731763325901154f * dz * (5.0f * z2 - 3.0f);
        sh[13] =  0.4570457994644657f * dx * (1.0f - 5.0f * z2);
        sh[14] =  1.445305721320277f * dz * (x2 - y2);
        sh[15] =  0.5900435899266435f * dx * (-x2 + 3.0f * y2);
#pragma unroll
        for (int i = 0; i < 16; ++i) er[16 + i] = (_Float16)sh[i];
    }
    __syncthreads();

    // ---- stage 2: MLP via WMMA; each wave owns a 16-row M slab ----
    const int wave = t >> 5;
    const int lane = t & 31;
    const int m0   = wave * 16;

    // layer 1: enc(32) -> actA(128), relu
    {
        const v16h a = load_frag_a(s_enc + m0 * ENCS, ENCS, lane);
#pragma unroll
        for (int nt = 0; nt < 8; ++nt) {
            const v16h bf = load_frag_b(s_w1 + (nt * 16) * ENCS, ENCS, 0, lane);
            v8f c = {};
            c = __builtin_amdgcn_wmma_f32_16x16x32_f16(false, a, false, bf,
                                                       (short)0, c, false, false);
            store_act_tile(s_actA, m0, nt * 16, c, s_b1, true, lane);
        }
    }
    __syncthreads();

    // layer 2: actA(128) -> actB(128), relu
    {
        v16h a[4];
#pragma unroll
        for (int kt = 0; kt < 4; ++kt)
            a[kt] = load_frag_a(s_actA + m0 * ACTS + kt * 32, ACTS, lane);
#pragma unroll
        for (int nt = 0; nt < 8; ++nt) {
            v8f c = {};
#pragma unroll
            for (int kt = 0; kt < 4; ++kt) {
                const v16h bf = load_frag_b(s_w2 + (nt * 16) * ACTS, ACTS, kt * 32, lane);
                c = __builtin_amdgcn_wmma_f32_16x16x32_f16(false, a[kt], false, bf,
                                                           (short)0, c, false, false);
            }
            store_act_tile(s_actB, m0, nt * 16, c, s_b2, true, lane);
        }
    }
    __syncthreads();

    // layer 3: actB(128) -> actA(128), relu
    {
        v16h a[4];
#pragma unroll
        for (int kt = 0; kt < 4; ++kt)
            a[kt] = load_frag_a(s_actB + m0 * ACTS + kt * 32, ACTS, lane);
#pragma unroll
        for (int nt = 0; nt < 8; ++nt) {
            v8f c = {};
#pragma unroll
            for (int kt = 0; kt < 4; ++kt) {
                const v16h bf = load_frag_b(s_w3 + (nt * 16) * ACTS, ACTS, kt * 32, lane);
                c = __builtin_amdgcn_wmma_f32_16x16x32_f16(false, a[kt], false, bf,
                                                           (short)0, c, false, false);
            }
            store_act_tile(s_actA, m0, nt * 16, c, s_b3, true, lane);
        }
    }
    __syncthreads();

    // ---- stage 3: output layer 128 -> 2 (too skinny for WMMA): f32 dots ----
    {
        const int p2 = t >> 1;     // local point
        const int o  = t & 1;      // output channel
        const _Float16* ar = s_actA + p2 * ACTS;
        const float*    wr = s_w4 + o * NEUR;
        float acc = s_b4[o];
#pragma unroll 8
        for (int k = 0; k < NEUR; ++k)
            acc = fmaf((float)ar[k], wr[k], acc);
        const long pgo = (long)blockIdx.x * PTS + p2;
        if (pgo < N) out[pgo * 2 + o] = acc;
    }
}

extern "C" void kernel_launch(void* const* d_in, const int* in_sizes, int n_in,
                              void* d_out, int out_size, void* d_ws, size_t ws_size,
                              hipStream_t stream) {
    (void)n_in; (void)d_ws; (void)ws_size; (void)out_size;
    const float* x     = (const float*)d_in[0];
    const float* table = (const float*)d_in[1];
    const float* W1    = (const float*)d_in[2];
    const float* b1    = (const float*)d_in[3];
    const float* W2    = (const float*)d_in[4];
    const float* b2    = (const float*)d_in[5];
    const float* W3    = (const float*)d_in[6];
    const float* b3    = (const float*)d_in[7];
    const float* W4    = (const float*)d_in[8];
    const float* b4    = (const float*)d_in[9];
    float* out = (float*)d_out;

    const int N = in_sizes[0] / 6;
    const int blocks = (N + PTS - 1) / PTS;
    hipLaunchKernelGGL(ngp_fused_mlp_kernel, dim3(blocks), dim3(THREADS), 0, stream,
                       x, table, W1, b1, W2, b2, W3, b3, W4, b4, out, N);
}
